// MultiHeadAttention_74105365725531
// MI455X (gfx1250) — matrix operand order repaired and checked by reading, not compiled, not run on hardware
//
#include <hip/hip_runtime.h>

#define BS 8
#define NPOS 384
#define DIN 512
#define DOUT 512
#define H 8
#define DK 64
#define RTOT (BS*NPOS)          /* 3072 */
#define INV_SQRT_DK 0.125f

typedef __attribute__((ext_vector_type(16))) __bf16       v16bf;
typedef __attribute__((ext_vector_type(4)))  __bf16       v4bf;
typedef __attribute__((ext_vector_type(8)))  float        v8f;
typedef __attribute__((ext_vector_type(4)))  unsigned int u32x4;

// ---- WMMA fragment helpers (layouts per CDNA5 ISA 7.12.2, wave32) ----------
// A: 16x32 bf16. lane 0-15 -> M=lane, K halves {0..7,16..23}; lane 16-31 -> M=lane-16,
// K halves {8..15,24..31}. Two 16-byte loads per lane.
static __device__ inline v16bf frag_a(const __bf16* base, int lda) {
  const int lane = threadIdx.x & 31;
  const int row  = lane & 15;
  const int k0   = (lane >> 4) << 3;             // 0 or 8
  const __bf16* p = base + row * lda + k0;
  union { u32x4 u[2]; v16bf v; } t;
  t.u[0] = *(const u32x4*)(p);                   // K = k0 .. k0+7
  t.u[1] = *(const u32x4*)(p + 16);              // K = k0+16 .. k0+23
  return t.v;
}
// B: 32x16 bf16. lane 0-15 -> N=lane, K {0..7,16..23}; lane 16-31 -> N=lane-16, K {8..15,24..31}.
static __device__ inline v16bf frag_b(const __bf16* base, int ldb) {
  const int lane = threadIdx.x & 31;
  const int k0   = (lane >> 4) << 3;
  const __bf16* p = base + (lane & 15);
  v16bf v;
#pragma unroll
  for (int i = 0; i < 8; ++i) {
    v[i]     = p[(size_t)(k0 + i) * ldb];
    v[8 + i] = p[(size_t)(k0 + 16 + i) * ldb];
  }
  return v;
}
static __device__ inline v8f wmma_bf16(v16bf a, v16bf b, v8f c) {
  return __builtin_amdgcn_wmma_f32_16x16x32_bf16(false, a, false, b, (short)0, c,
                                                 false, false);
}

// ---------------------------------------------------------------------------
// Kernel 1: fused Q/K/V projections.  X(3072x512) @ W(512x512) + b.
// mode 0: -> Qb bf16 (token-major)        mode 1: -> Kbt bf16 [b][h][d][m]
// mode 2: -> original_v f32 (d_out) and Vb bf16 (token-major)
// Block: 64x64 output tile, 128 threads = 4 waves, each wave a 16x64 strip.
// ---------------------------------------------------------------------------
__global__ void __launch_bounds__(128)
k_qkv_proj(const float* __restrict__ q, const float* __restrict__ k,
           const float* __restrict__ v,
           const float* __restrict__ Wq, const float* __restrict__ bq,
           const float* __restrict__ Wk, const float* __restrict__ bk,
           const float* __restrict__ Wv, const float* __restrict__ bv,
           __bf16* __restrict__ Qb, __bf16* __restrict__ Kbt,
           __bf16* __restrict__ Vb, float* __restrict__ Vout)
{
  const int mode = blockIdx.z;
  const float* X    = mode == 0 ? q  : (mode == 1 ? k  : v);
  const float* W    = mode == 0 ? Wq : (mode == 1 ? Wk : Wv);
  const float* bias = mode == 0 ? bq : (mode == 1 ? bk : bv);
  const int row0 = blockIdx.x * 64;
  const int col0 = blockIdx.y * 64;
  const int wave = threadIdx.x >> 5;
  const int lane = threadIdx.x & 31;

  __shared__ __align__(16) __bf16 Xs[64][40];   // lda 40 -> 80B rows (16B aligned)
  __shared__ __align__(16) __bf16 Ws[32][80];   // ldb 80 -> 160B rows

  v8f acc[4] = {};

  for (int kt = 0; kt < DIN / 32; ++kt) {
    __syncthreads();
    // X chunk: 64 rows x 32 k, float4 loads + packed v4bf LDS stores
    for (int i = threadIdx.x; i < 64 * 8; i += 128) {
      int r = i >> 3, cq = i & 7;
      const float* src = &X[(size_t)(row0 + r) * DIN + kt * 32 + cq * 4];
      const float4 f = *(const float4*)src;
      if (kt + 1 < DIN / 32) __builtin_prefetch(src + 32, 0, 1);
      v4bf p = { (__bf16)f.x, (__bf16)f.y, (__bf16)f.z, (__bf16)f.w };
      *(v4bf*)&Xs[r][cq * 4] = p;
    }
    // W chunk: 32 k x 64 cols
    for (int i = threadIdx.x; i < 32 * 16; i += 128) {
      int r = i >> 4, cq = i & 15;
      const float4 f = *(const float4*)&W[(size_t)(kt * 32 + r) * DOUT + col0 + cq * 4];
      v4bf p = { (__bf16)f.x, (__bf16)f.y, (__bf16)f.z, (__bf16)f.w };
      *(v4bf*)&Ws[r][cq * 4] = p;
    }
    __syncthreads();
    v16bf a = frag_a(&Xs[wave * 16][0], 40);
#pragma unroll
    for (int s = 0; s < 4; ++s) {
      v16bf bb = frag_b(&Ws[0][s * 16], 80);
      acc[s] = wmma_bf16(a, bb, acc[s]);
    }
  }

  const int rhi = (lane >> 4) * 8;
#pragma unroll
  for (int s = 0; s < 4; ++s) {
    const int col = col0 + s * 16 + (lane & 15);
    const float bval = bias[col];
#pragma unroll
    for (int r = 0; r < 8; ++r) {
      const int grow = row0 + wave * 16 + rhi + r;
      const float val = acc[s][r] + bval;
      if (mode == 0) {
        Qb[(size_t)grow * DOUT + col] = (__bf16)val;
      } else if (mode == 1) {
        const int bb2 = grow / NPOS, m = grow % NPOS;
        const int hh = col >> 6, d = col & 63;
        Kbt[(((size_t)bb2 * H + hh) * DK + d) * NPOS + m] = (__bf16)val;
      } else {
        Vout[(size_t)grow * DOUT + col] = val;      // original_v output
        Vb[(size_t)grow * DOUT + col]   = (__bf16)val;
      }
    }
  }
}

// ---------------------------------------------------------------------------
// Kernel 2: relative-position scores, rpos_k read EXACTLY ONCE (302MB stream,
// the roofline-dominant traffic -> b128 loads).
// For each (n,h): Pos[b][m] = Q[b,h,n,:] (8x64, padded to 16) @ rpos_k[n,:,h,:]^T.
// rpos slice staged transposed to LDS as bf16; 4 waves cover 24 m-tiles.
// ---------------------------------------------------------------------------
__global__ void __launch_bounds__(128)
k_pos(const float* __restrict__ rpos_k, const __bf16* __restrict__ Qb,
      float* __restrict__ Pos)
{
  const int n = blockIdx.x;
  const int h = blockIdx.y;
  const int wave = threadIdx.x >> 5;
  const int lane = threadIdx.x & 31;

  __shared__ __align__(16) __bf16 Bt[DK][NPOS + 8];  // [d][m], ld 392 (784B rows)
  __shared__ __align__(16) __bf16 Qs[16][DK];

  // 384 m x 64 d slice: float4 (b128) loads, transposed bf16 stores
  for (int i = threadIdx.x; i < NPOS * (DK / 4); i += 128) {
    int m = i >> 4, dq = i & 15;
    const float4 f =
        *(const float4*)&rpos_k[((size_t)n * NPOS + m) * (H * DK) + h * DK + dq * 4];
    Bt[dq * 4 + 0][m] = (__bf16)f.x;
    Bt[dq * 4 + 1][m] = (__bf16)f.y;
    Bt[dq * 4 + 2][m] = (__bf16)f.z;
    Bt[dq * 4 + 3][m] = (__bf16)f.w;
  }
  // Q rows for all 8 batches (rows 8..15 zero), b128 copies
  for (int i = threadIdx.x; i < 16 * 8; i += 128) {
    int r = i >> 3, cq = i & 7;
    u32x4 z = {0u, 0u, 0u, 0u};
    if (r < BS)
      z = *(const u32x4*)&Qb[((size_t)r * NPOS + n) * DOUT + h * DK + cq * 8];
    *(u32x4*)&Qs[r][cq * 8] = z;
  }
  __syncthreads();

  for (int mt = wave; mt < NPOS / 16; mt += 4) {
    v8f c = {};
#pragma unroll
    for (int kk = 0; kk < 2; ++kk) {
      v16bf a  = frag_a(&Qs[0][kk * 32], DK);
      v16bf bb = frag_b(&Bt[kk * 32][mt * 16], NPOS + 8);
      c = wmma_bf16(a, bb, c);
    }
    const int rhi = (lane >> 4) * 8;
    const int col = mt * 16 + (lane & 15);
#pragma unroll
    for (int r = 0; r < 8; ++r) {
      const int bb2 = rhi + r;
      if (bb2 < BS)
        Pos[(((size_t)bb2 * H + h) * NPOS + n) * NPOS + col] = c[r];
    }
  }
}

// ---------------------------------------------------------------------------
// Kernel 3: attention per (b, h, 16-row tile of n).
// scores (WMMA, Kbt B-operand contiguous) + pos, softmax (wave-per-row,
// wave32 shuffle reductions, 1/sqrt(dk) folded into normalization),
// then scores @ V (WMMA over 12 K-chunks of the 384 m dimension).
// ---------------------------------------------------------------------------
__global__ void __launch_bounds__(128)
k_attn(const __bf16* __restrict__ Qb, const __bf16* __restrict__ Kbt,
       const __bf16* __restrict__ Vb, const float* __restrict__ Pos,
       __bf16* __restrict__ Ab)
{
  const int nt = blockIdx.x;          // 0..23
  const int h  = blockIdx.y;
  const int b  = blockIdx.z;
  const int n0 = nt * 16;
  const int wave = threadIdx.x >> 5;
  const int lane = threadIdx.x & 31;

  __shared__ __align__(16) __bf16 Qs[16][DK];
  __shared__ __align__(16) float  sS[16][NPOS + 8];
  __shared__ __align__(16) __bf16 sP[16][NPOS + 8];

  for (int i = threadIdx.x; i < 16 * 8; i += 128) {
    int r = i >> 3, cq = i & 7;
    *(u32x4*)&Qs[r][cq * 8] =
        *(const u32x4*)&Qb[((size_t)b * NPOS + n0 + r) * DOUT + h * DK + cq * 8];
  }
  __syncthreads();

  // ---- content scores + pos scores, scaled ----
  const __bf16* kbase = Kbt + ((size_t)b * H + h) * DK * NPOS;
  const float*  prow  = Pos + (((size_t)b * H + h) * NPOS + n0) * NPOS;
  for (int mt = wave; mt < NPOS / 16; mt += 4) {
    v8f c = {};
#pragma unroll
    for (int kk = 0; kk < 2; ++kk) {
      v16bf a  = frag_a(&Qs[0][kk * 32], DK);
      v16bf bb = frag_b(kbase + (size_t)(kk * 32) * NPOS + mt * 16, NPOS);
      c = wmma_bf16(a, bb, c);
    }
    const int rhi = (lane >> 4) * 8;
    const int col = mt * 16 + (lane & 15);
#pragma unroll
    for (int r = 0; r < 8; ++r) {
      const int row = rhi + r;
      sS[row][col] = (c[r] + prow[(size_t)row * NPOS + col]) * INV_SQRT_DK;
    }
  }
  __syncthreads();

  // ---- softmax, one wave32 per row; fold trailing 1/sqrt(dk) ----
  for (int row = wave; row < 16; row += 4) {
    float vals[12];
    float mx = -3.0e38f;
#pragma unroll
    for (int j = 0; j < 12; ++j) {
      vals[j] = sS[row][lane + j * 32];
      mx = fmaxf(mx, vals[j]);
    }
#pragma unroll
    for (int off = 16; off > 0; off >>= 1) mx = fmaxf(mx, __shfl_xor(mx, off, 32));
    float sum = 0.f;
#pragma unroll
    for (int j = 0; j < 12; ++j) { vals[j] = __expf(vals[j] - mx); sum += vals[j]; }
#pragma unroll
    for (int off = 16; off > 0; off >>= 1) sum += __shfl_xor(sum, off, 32);
    const float sc = INV_SQRT_DK / sum;
#pragma unroll
    for (int j = 0; j < 12; ++j) sP[row][lane + j * 32] = (__bf16)(vals[j] * sc);
  }
  __syncthreads();

  // ---- attn_out = probs @ V : wave w computes d-subtile [w*16, w*16+16) ----
  const __bf16* vbase = Vb + (size_t)b * NPOS * DOUT + h * DK + wave * 16;
  v8f c = {};
  for (int kk = 0; kk < NPOS / 32; ++kk) {
    v16bf a  = frag_a(&sP[0][kk * 32], NPOS + 8);
    v16bf bb = frag_b(vbase + (size_t)(kk * 32) * DOUT, DOUT);
    c = wmma_bf16(a, bb, c);
  }
  const int rhi = (lane >> 4) * 8;
  const int col = lane & 15;
#pragma unroll
  for (int r = 0; r < 8; ++r) {
    const int row = rhi + r;
    Ab[((size_t)b * NPOS + n0 + row) * DOUT + h * DK + wave * 16 + col] = (__bf16)c[r];
  }
}

// ---------------------------------------------------------------------------
// Kernel 4: output projection  out = concat(3072x512 bf16) @ Wo + bo  (f32 out)
// ---------------------------------------------------------------------------
__global__ void __launch_bounds__(128)
k_oproj(const __bf16* __restrict__ Ab, const float* __restrict__ Wo,
        const float* __restrict__ bo, float* __restrict__ Out)
{
  const int row0 = blockIdx.x * 64;
  const int col0 = blockIdx.y * 64;
  const int wave = threadIdx.x >> 5;
  const int lane = threadIdx.x & 31;

  __shared__ __align__(16) __bf16 Xs[64][40];
  __shared__ __align__(16) __bf16 Ws[32][80];

  v8f acc[4] = {};
  for (int kt = 0; kt < DOUT / 32; ++kt) {
    __syncthreads();
    // bf16 activations: straight b128 copies
    for (int i = threadIdx.x; i < 64 * 4; i += 128) {
      int r = i >> 2, cq = i & 3;
      const __bf16* src = &Ab[(size_t)(row0 + r) * DOUT + kt * 32 + cq * 8];
      if (kt + 1 < DOUT / 32) __builtin_prefetch(src + 32, 0, 1);
      *(u32x4*)&Xs[r][cq * 8] = *(const u32x4*)src;
    }
    for (int i = threadIdx.x; i < 32 * 16; i += 128) {
      int r = i >> 4, cq = i & 15;
      const float4 f = *(const float4*)&Wo[(size_t)(kt * 32 + r) * DOUT + col0 + cq * 4];
      v4bf p = { (__bf16)f.x, (__bf16)f.y, (__bf16)f.z, (__bf16)f.w };
      *(v4bf*)&Ws[r][cq * 4] = p;
    }
    __syncthreads();
    v16bf a = frag_a(&Xs[wave * 16][0], 40);
#pragma unroll
    for (int s = 0; s < 4; ++s) {
      v16bf bb = frag_b(&Ws[0][s * 16], 80);
      acc[s] = wmma_bf16(a, bb, acc[s]);
    }
  }
  const int rhi = (lane >> 4) * 8;
#pragma unroll
  for (int s = 0; s < 4; ++s) {
    const int col = col0 + s * 16 + (lane & 15);
    const float bval = bo[col];
#pragma unroll
    for (int r = 0; r < 8; ++r) {
      const int grow = row0 + wave * 16 + rhi + r;
      Out[(size_t)grow * DOUT + col] = acc[s][r] + bval;
    }
  }
}

// ---------------------------------------------------------------------------
extern "C" void kernel_launch(void* const* d_in, const int* in_sizes, int n_in,
                              void* d_out, int out_size, void* d_ws, size_t ws_size,
                              hipStream_t stream) {
  (void)in_sizes; (void)n_in; (void)out_size; (void)ws_size;
  const float* q      = (const float*)d_in[0];
  const float* k      = (const float*)d_in[1];
  const float* v      = (const float*)d_in[2];
  const float* rpos_k = (const float*)d_in[3];
  /* d_in[4] = rpos_v : unused by reference */
  const float* Wq = (const float*)d_in[5];
  const float* bq = (const float*)d_in[6];
  const float* Wk = (const float*)d_in[7];
  const float* bk = (const float*)d_in[8];
  const float* Wv = (const float*)d_in[9];
  const float* bv = (const float*)d_in[10];
  const float* Wo = (const float*)d_in[11];
  const float* bo = (const float*)d_in[12];

  float* out_v = (float*)d_out;                         // original_v (first tuple elem)
  float* out_y = (float*)d_out + (size_t)RTOT * DOUT;   // final output

  char* ws = (char*)d_ws;
  __bf16* Qb  = (__bf16*)ws;  ws += (size_t)RTOT * DOUT * 2;   // 3.0 MB
  __bf16* Kbt = (__bf16*)ws;  ws += (size_t)RTOT * DOUT * 2;   // 3.0 MB [b][h][d][m]
  __bf16* Vb  = (__bf16*)ws;  ws += (size_t)RTOT * DOUT * 2;   // 3.0 MB
  float*  Pos = (float*)ws;   ws += (size_t)BS * H * NPOS * NPOS * 4; // 36 MB
  __bf16* Ab  = (__bf16*)ws;                                    // 3.0 MB

  dim3 blk(128);
  k_qkv_proj<<<dim3(RTOT / 64, DOUT / 64, 3), blk, 0, stream>>>(
      q, k, v, Wq, bq, Wk, bk, Wv, bv, Qb, Kbt, Vb, out_v);
  k_pos<<<dim3(NPOS, H), blk, 0, stream>>>(rpos_k, Qb, Pos);
  k_attn<<<dim3(NPOS / 16, H, BS), blk, 0, stream>>>(Qb, Kbt, Vb, Pos, Ab);
  k_oproj<<<dim3(RTOT / 64, DOUT / 64), blk, 0, stream>>>(Ab, Wo, bo, out_y);
}
